// Model_39676907882464
// MI455X (gfx1250) — matrix operand order repaired and checked by reading, not compiled, not run on hardware
//
#include <hip/hip_runtime.h>
#include <hip/hip_bf16.h>

typedef __attribute__((ext_vector_type(16))) __bf16 v16bf;
typedef __attribute__((ext_vector_type(8)))  float  v8f;

#define WMMA_BF16(a, b, c) \
  __builtin_amdgcn_wmma_f32_16x16x32_bf16(false, (a), false, (b), (short)0, (c), false, false)

// ---------------------------------------------------------------------------
// Operand loaders for v_wmma_f32_16x16x32_bf16 (wave32 layouts per CDNA5 ISA)
//
// A (16x32, MxK): lane L: M = L&15; lanes<16 hold K=0..7,16..23; lanes>=16
//                 hold K=8..15,24..31 (2 bf16 per VGPR, 8 VGPRs).
// B (32x16, KxN): lane L: N = L&15; lanes<16 hold K=0..7,16..23; lanes>=16 K=8..15,24..31.
// C (16x16 f32):  VGPR r: lanes<16 -> M=r, N=lane; lanes>=16 -> M=8+r.
// ---------------------------------------------------------------------------

static __device__ inline v16bf load_A_f32(const float* A, int lda, int lane) {
  const int m  = lane & 15;
  const int kb = (lane >> 4) << 3;               // 0 or 8
  const float* p = A + (size_t)m * lda + kb;
  v16bf a;
#pragma unroll
  for (int j = 0; j < 8; ++j) a[j]     = (__bf16)p[j];
#pragma unroll
  for (int j = 0; j < 8; ++j) a[8 + j] = (__bf16)p[16 + j];
  return a;
}

static __device__ inline v16bf load_A_bf16(const __bf16* A, int lda, int lane) {
  const int m  = lane & 15;
  const int kb = (lane >> 4) << 3;
  const __bf16* p = A + (size_t)m * lda + kb;
  v16bf a;
#pragma unroll
  for (int j = 0; j < 8; ++j) a[j]     = p[j];
#pragma unroll
  for (int j = 0; j < 8; ++j) a[8 + j] = p[16 + j];
  return a;
}

// Source stores B transposed: row n of src == column n of B, contiguous in K.
static __device__ inline v16bf load_Bt_f32(const float* S, int ld, int lane) {
  const int n  = lane & 15;
  const int kb = (lane >> 4) << 3;               // 0 or 8
  const float* p = S + (size_t)n * ld + kb;
  v16bf b;
#pragma unroll
  for (int j = 0; j < 8; ++j) { b[j] = (__bf16)p[j]; b[8 + j] = (__bf16)p[16 + j]; }
  return b;
}

static __device__ inline v16bf load_Bt_bf16(const __bf16* S, int ld, int lane) {
  const int n  = lane & 15;
  const int kb = (lane >> 4) << 3;
  typedef __attribute__((ext_vector_type(8))) __bf16 v8bf;
  const __bf16* p = S + (size_t)n * ld + kb;
  v8bf lo = *(const v8bf*)p;
  v8bf hi = *(const v8bf*)(p + 16);
  return __builtin_shufflevector(lo, hi, 0,1,2,3,4,5,6,7,8,9,10,11,12,13,14,15);
}

// ---------------------------------------------------------------------------
// Kernel 1: C[m][n] = sum_k A[m][k] * W[n][k]  (fp32 in, bf16 out)
// Each wave computes a 16x64 strip (A reused across 4 B tiles).
// TRANSPOSED: write out[b][n][m&2047] with b = m>>11 (batch_rows = 2048).
// ---------------------------------------------------------------------------
template <int K, bool TRANSPOSED>
__global__ __launch_bounds__(32, 1) void proj_kernel(
    const float* __restrict__ A, const float* __restrict__ W,
    __bf16* __restrict__ out, int N) {
  const int lane = threadIdx.x;
  const int m0 = blockIdx.x * 16;
  const int n0 = blockIdx.y * 64;
  v8f acc[4];
#pragma unroll
  for (int t = 0; t < 4; ++t)
#pragma unroll
    for (int r = 0; r < 8; ++r) acc[t][r] = 0.f;

#pragma unroll 2
  for (int k = 0; k < K; k += 32) {
    v16bf a = load_A_f32(A + (size_t)m0 * K + k, K, lane);
#pragma unroll
    for (int t = 0; t < 4; ++t) {
      v16bf bm = load_Bt_f32(W + (size_t)(n0 + 16 * t) * K + k, K, lane);
      acc[t] = WMMA_BF16(a, bm, acc[t]);
    }
  }
  const int n  = lane & 15;
  const int mb = (lane >> 4) << 3;
#pragma unroll
  for (int t = 0; t < 4; ++t)
#pragma unroll
    for (int r = 0; r < 8; ++r) {
      const int m = m0 + mb + r;
      if (!TRANSPOSED) {
        out[(size_t)m * N + n0 + 16 * t + n] = (__bf16)acc[t][r];
      } else {
        out[((size_t)(m >> 11) * N + n0 + 16 * t + n) * 2048 + (m & 2047)] =
            (__bf16)acc[t][r];
      }
    }
}

// ---------------------------------------------------------------------------
// Kernel 2: valT[b][v][k] = (bf16) value[b][k][v]  via LDS 32x32 tile transpose
// block (32,8), grid (2048/32, 256/32, 8). Coalesced reads and writes.
// ---------------------------------------------------------------------------
__global__ __launch_bounds__(256) void transpose_val(
    const float* __restrict__ val, __bf16* __restrict__ valT) {
  __shared__ float t[32][33];
  const int b  = blockIdx.z;
  const int k0 = blockIdx.x * 32;
  const int v0 = blockIdx.y * 32;
  const int tx = threadIdx.x, ty = threadIdx.y;
#pragma unroll
  for (int i = 0; i < 32; i += 8)
    t[ty + i][tx] = val[((size_t)b * 2048 + k0 + ty + i) * 256 + v0 + tx];
  __syncthreads();
#pragma unroll
  for (int i = 0; i < 32; i += 8)
    valT[((size_t)b * 256 + v0 + ty + i) * 2048 + k0 + tx] = (__bf16)t[tx][ty + i];
}

// ---------------------------------------------------------------------------
// Kernel 3: flash attention.  One wave = one (b, 16-row q tile).
//   S = (r1_tile @ r2^T) * 4.5 ; online softmax over Sk ; O += P @ r3
//   writes v1[b][q][o] bf16.
// ---------------------------------------------------------------------------
__global__ __launch_bounds__(32, 1) void attn_kernel(
    const __bf16* __restrict__ r1, const __bf16* __restrict__ r2,
    const __bf16* __restrict__ r3t, __bf16* __restrict__ v1) {
  const int lane = threadIdx.x;
  const int b  = blockIdx.y;             // 0..7
  const int q0 = blockIdx.x * 16;        // 0..2032
  __shared__ float plds[16 * 32];

  // Resident A tile: 16 q-rows x 128 d  (4 chunks of K=32)
  v16bf aq[4];
#pragma unroll
  for (int i = 0; i < 4; ++i)
    aq[i] = load_A_bf16(r1 + ((size_t)b * 2048 + q0) * 128 + 32 * i, 128, lane);

  float mrun[8], lrun[8];
  v8f oacc[8];
#pragma unroll
  for (int r = 0; r < 8; ++r) { mrun[r] = -3.0e38f; lrun[r] = 0.f; }
#pragma unroll
  for (int j = 0; j < 8; ++j)
#pragma unroll
    for (int r = 0; r < 8; ++r) oacc[j][r] = 0.f;

  for (int kt = 0; kt < 2048; kt += 32) {
    v8f s0 = {0.f, 0.f, 0.f, 0.f, 0.f, 0.f, 0.f, 0.f};
    v8f s1 = {0.f, 0.f, 0.f, 0.f, 0.f, 0.f, 0.f, 0.f};
#pragma unroll
    for (int i = 0; i < 4; ++i) {
      v16bf b0 = load_Bt_bf16(r2 + ((size_t)b * 2048 + kt) * 128 + 32 * i, 128, lane);
      v16bf b1 = load_Bt_bf16(r2 + ((size_t)b * 2048 + kt + 16) * 128 + 32 * i, 128, lane);
      s0 = WMMA_BF16(aq[i], b0, s0);
      s1 = WMMA_BF16(aq[i], b1, s1);
    }

    // scale by 4.5 (qk / inv_scale), online softmax over the 32 columns
    float corr[8];
#pragma unroll
    for (int r = 0; r < 8; ++r) {
      float x0 = s0[r] * 4.5f, x1 = s1[r] * 4.5f;
      float mx = fmaxf(x0, x1);
      mx = fmaxf(mx, __shfl_xor(mx, 1));
      mx = fmaxf(mx, __shfl_xor(mx, 2));
      mx = fmaxf(mx, __shfl_xor(mx, 4));
      mx = fmaxf(mx, __shfl_xor(mx, 8));   // reduce within 16-lane half = one row
      const float mnew = fmaxf(mrun[r], mx);
      corr[r] = __expf(mrun[r] - mnew);
      const float e0 = __expf(x0 - mnew);
      const float e1 = __expf(x1 - mnew);
      s0[r] = e0; s1[r] = e1;
      float rs = e0 + e1;
      rs += __shfl_xor(rs, 1);
      rs += __shfl_xor(rs, 2);
      rs += __shfl_xor(rs, 4);
      rs += __shfl_xor(rs, 8);
      lrun[r] = lrun[r] * corr[r] + rs;
      mrun[r] = mnew;
    }
#pragma unroll
    for (int j = 0; j < 8; ++j)
#pragma unroll
      for (int r = 0; r < 8; ++r) oacc[j][r] *= corr[r];

    // Re-layout P: C-layout f32 -> LDS -> A-layout bf16
    {
      const int col  = lane & 15;
      const int rowb = (lane >> 4) << 3;
#pragma unroll
      for (int r = 0; r < 8; ++r) {
        plds[(rowb + r) * 32 + col]      = s0[r];
        plds[(rowb + r) * 32 + 16 + col] = s1[r];
      }
    }
    __syncthreads();
    v16bf p;
    {
      const int m  = lane & 15;
      const int kb = (lane >> 4) << 3;
      const float* pp = plds + m * 32 + kb;
#pragma unroll
      for (int j = 0; j < 8; ++j) { p[j] = (__bf16)pp[j]; p[8 + j] = (__bf16)pp[16 + j]; }
    }
    __syncthreads();

    // O(16x128) += P(16x32) @ r3_chunk(32x128)
#pragma unroll
    for (int j = 0; j < 8; ++j) {
      v16bf bv = load_Bt_bf16(r3t + ((size_t)b * 128 + j * 16) * 2048 + kt, 2048, lane);
      oacc[j] = WMMA_BF16(p, bv, oacc[j]);
    }
  }

  // normalize and store v1 (bf16)
  const int n  = lane & 15;
  const int mb = (lane >> 4) << 3;
#pragma unroll
  for (int j = 0; j < 8; ++j)
#pragma unroll
    for (int r = 0; r < 8; ++r) {
      const float o = oacc[j][r] / lrun[r];
      v1[((size_t)b * 2048 + q0 + mb + r) * 128 + j * 16 + n] = (__bf16)o;
    }
}

// ---------------------------------------------------------------------------
// Kernel 4: out[b][q][v] = v1[b][q][:] @ Wout[v][:] + mask[b][q][:] @ value[:,v]
// Each wave computes a 16x64 strip: the mask A tile (the 128 MB operand) is
// converted once and reused across 4 WMMAs.
// ---------------------------------------------------------------------------
__global__ __launch_bounds__(32, 1) void out_kernel(
    const __bf16* __restrict__ v1, const float* __restrict__ Wout,
    const float* __restrict__ mask, const __bf16* __restrict__ valT,
    float* __restrict__ out) {
  const int lane = threadIdx.x;
  const int m0 = blockIdx.x * 16;       // flattened b*2048+q, 0..16383
  const int n0 = blockIdx.y * 64;       // v, 0..255 in strips of 64
  const int b  = m0 >> 11;
  v8f acc[4];
#pragma unroll
  for (int t = 0; t < 4; ++t)
#pragma unroll
    for (int r = 0; r < 8; ++r) acc[t][r] = 0.f;

  // v1 @ Wout^T, K = 128
#pragma unroll
  for (int kc = 0; kc < 128; kc += 32) {
    v16bf a = load_A_bf16(v1 + (size_t)m0 * 128 + kc, 128, lane);
#pragma unroll
    for (int t = 0; t < 4; ++t) {
      v16bf w = load_Bt_f32(Wout + (size_t)(n0 + 16 * t) * 128 + kc, 128, lane);
      acc[t] = WMMA_BF16(a, w, acc[t]);
    }
  }
  // mask @ value, K = 2048 (valT is value transposed, bf16)
#pragma unroll 2
  for (int kc = 0; kc < 2048; kc += 32) {
    v16bf a = load_A_f32(mask + (size_t)m0 * 2048 + kc, 2048, lane);
#pragma unroll
    for (int t = 0; t < 4; ++t) {
      v16bf w = load_Bt_bf16(valT + ((size_t)b * 256 + n0 + 16 * t) * 2048 + kc, 2048, lane);
      acc[t] = WMMA_BF16(a, w, acc[t]);
    }
  }
  const int n  = lane & 15;
  const int mb = (lane >> 4) << 3;
#pragma unroll
  for (int t = 0; t < 4; ++t)
#pragma unroll
    for (int r = 0; r < 8; ++r)
      out[(size_t)(m0 + mb + r) * 256 + n0 + 16 * t + n] = acc[t][r];
}

// ---------------------------------------------------------------------------
extern "C" void kernel_launch(void* const* d_in, const int* in_sizes, int n_in,
                              void* d_out, int out_size, void* d_ws, size_t ws_size,
                              hipStream_t stream) {
  (void)in_sizes; (void)n_in; (void)out_size; (void)ws_size;
  const float* query = (const float*)d_in[0];
  const float* key   = (const float*)d_in[1];
  const float* value = (const float*)d_in[2];
  const float* mask  = (const float*)d_in[3];
  const float* W0    = (const float*)d_in[4];
  const float* W1    = (const float*)d_in[5];
  const float* W2    = (const float*)d_in[6];
  const float* Wout  = (const float*)d_in[7];
  float* out = (float*)d_out;

  char* w = (char*)d_ws;
  __bf16* r1   = (__bf16*)w;  w += (size_t)16384 * 128  * 2;   // [b*2048+q][128]
  __bf16* r2   = (__bf16*)w;  w += (size_t)16384 * 128  * 2;   // [b*2048+k][128]
  __bf16* r3t  = (__bf16*)w;  w += (size_t)8 * 128 * 2048 * 2; // [b][o][k]
  __bf16* valT = (__bf16*)w;  w += (size_t)8 * 256 * 2048 * 2; // [b][v][k]
  __bf16* v1   = (__bf16*)w;  w += (size_t)16384 * 128  * 2;   // [b*2048+q][128]

  // r1 = query @ W0^T, r2 = key @ W1^T   (M=16384, N=128, K=128)
  proj_kernel<128, false><<<dim3(1024, 2), 32, 0, stream>>>(query, W0, r1, 128);
  proj_kernel<128, false><<<dim3(1024, 2), 32, 0, stream>>>(key,   W1, r2, 128);
  // r3t = (value @ W2^T) transposed per batch  (M=16384, N=128, K=256)
  proj_kernel<256, true><<<dim3(1024, 2), 32, 0, stream>>>(value, W2, r3t, 128);
  // valT = value^T (bf16)
  transpose_val<<<dim3(64, 8, 8), dim3(32, 8), 0, stream>>>(value, valT);
  // flash attention -> v1
  attn_kernel<<<dim3(128, 8), 32, 0, stream>>>(r1, r2, r3t, v1);
  // out = v1 @ Wout^T + mask @ value
  out_kernel<<<dim3(1024, 4), 32, 0, stream>>>(v1, Wout, mask, valT, out);
}
